// GatedModularNet_1726576854117
// MI455X (gfx1250) — matrix operand order repaired and checked by reading, not compiled, not run on hardware
//
#include <hip/hip_runtime.h>
#include <cstdint>
#include <cstddef>

// Problem constants (match reference)
#define L_   4
#define M_   6
#define DIN  512
#define DD   1024
#define T_   8
#define DOUT 256
#define BB   1024

// GEMM tiling: 256 threads = 8 wave32; block tile 128x128, wave tile 32x64.
#define BM 128
#define BN 128
#define BK 32
#define KSTR 40   // ushorts per LDS plane row (80 B: 16B-aligned, bank-friendly)

typedef __attribute__((ext_vector_type(16))) __bf16          v16bf;
typedef __attribute__((ext_vector_type(16))) unsigned short  v16us;
typedef __attribute__((ext_vector_type(8)))  unsigned short  u16x8;
typedef __attribute__((ext_vector_type(4)))  unsigned short  u16x4;
typedef __attribute__((ext_vector_type(8)))  float           v8f;

// bf16 split: hi = truncate-to-bf16(x), lo = bf16(x - hi). hi*hi+hi*lo+lo*hi
// recovers ~f32 precision with f32 WMMA accumulation.
__device__ __forceinline__ unsigned short bf_hi(float x) {
    return (unsigned short)(__float_as_uint(x) >> 16);
}
__device__ __forceinline__ unsigned short bf_lo(float x) {
    float hi = __uint_as_float(__float_as_uint(x) & 0xffff0000u);
    return (unsigned short)(__float_as_uint(x - hi) >> 16);
}

// Fragment = two contiguous 16B LDS loads -> 8 packed-bf16 VGPRs, no shuffles.
__device__ __forceinline__ v16bf load_frag(const unsigned short* p0,
                                           const unsigned short* p1) {
    u16x8 a = *(const u16x8*)p0;
    u16x8 b = *(const u16x8*)p1;
    v16us w = __builtin_shufflevector(a, b, 0,1,2,3,4,5,6,7,8,9,10,11,12,13,14,15);
    return __builtin_bit_cast(v16bf, w);
}

#define WMMA_BF16(AH, BH, CC) \
    __builtin_amdgcn_wmma_f32_16x16x32_bf16(false, AH, false, BH, (short)0, CC, false, false)

// C[z] = act(A[z] (BxK) @ W[z] (KxN) + bias[z]), bf16x3 emulated f32 GEMM.
__global__ __launch_bounds__(256)
void gemm_bf16x3(const float* __restrict__ Abase, size_t Astride,
                 const float* __restrict__ Wbase, size_t Wstride,
                 const float* __restrict__ biasBase, size_t biasStride,
                 float* __restrict__ Cbase, size_t Cstride,
                 int K, int N, int do_relu)
{
    // Separate hi/lo bf16 planes; A: [row][k], B: [n][k] with 16B-chunk XOR swizzle.
    __shared__ __align__(16) unsigned short Ah[BM][KSTR];
    __shared__ __align__(16) unsigned short Al[BM][KSTR];
    __shared__ __align__(16) unsigned short Bh[BN][KSTR];
    __shared__ __align__(16) unsigned short Bl[BN][KSTR];

    const float* A    = Abase    + (size_t)blockIdx.z * Astride;
    const float* W    = Wbase    + (size_t)blockIdx.z * Wstride;
    const float* bias = biasBase + (size_t)blockIdx.z * biasStride;
    float*       C    = Cbase    + (size_t)blockIdx.z * Cstride;

    const int bRow = blockIdx.x * BM;
    const int bCol = blockIdx.y * BN;

    const int tid  = threadIdx.x;
    const int lane = tid & 31;
    const int wv   = tid >> 5;     // 8 waves
    const int h    = lane >> 4;    // K-half select (ISA fragment layout)
    const int lr   = lane & 15;    // row (A) / col (B,C) within 16
    const int rg   = wv >> 1;      // wave row group: rows rg*32 .. rg*32+31
    const int cg   = wv & 1;       // wave col group: cols cg*64 .. cg*64+63

    // B staging assignment: thread owns a 4k x 4n block at (4*kq, 4*c5)
    const int kq = tid >> 5;       // 0..7
    const int c5 = tid & 31;       // 0..31

    v8f acc[2][4] = {};

    for (int k0 = 0; k0 < K; k0 += BK) {
        // ---- stage A tile (BM x BK): coalesced float4, vectorized b64 stores ----
        #pragma unroll
        for (int i = 0; i < 4; ++i) {
            int idx = tid + (i << 8);       // 1024 float4
            int row = idx >> 3;             // 8 float4 per row
            int c4  = idx & 7;
            const float* ap = A + (size_t)(bRow + row) * K + k0 + c4 * 4;
            const float4 v = *(const float4*)ap;
            if (k0 + BK < K) __builtin_prefetch(ap + BK, 0, 1);
            u16x4 hi4 = { bf_hi(v.x), bf_hi(v.y), bf_hi(v.z), bf_hi(v.w) };
            u16x4 lo4 = { bf_lo(v.x), bf_lo(v.y), bf_lo(v.z), bf_lo(v.w) };
            *(u16x4*)&Ah[row][c4 * 4] = hi4;
            *(u16x4*)&Al[row][c4 * 4] = lo4;
        }
        // ---- stage B tile (BK x BN): coalesced loads, 4x4 register transpose,
        //      swizzled b64 stores along k (bank-conflict free) ----
        {
            float4 vv[4];
            #pragma unroll
            for (int r = 0; r < 4; ++r) {
                const float* wp = W + (size_t)(k0 + 4 * kq + r) * N + bCol + c5 * 4;
                vv[r] = *(const float4*)wp;
                if (k0 + BK < K) __builtin_prefetch(wp + (size_t)BK * N, 0, 1);
            }
            const float* vf = (const float*)vv;
            #pragma unroll
            for (int j = 0; j < 4; ++j) {
                int n = c5 * 4 + j;
                int f = ((n >> 4) ^ (n >> 6)) & 3;
                int koff = (((kq >> 1) ^ f) << 3) + ((kq & 1) << 2);
                u16x4 hi4, lo4;
                #pragma unroll
                for (int r = 0; r < 4; ++r) {
                    float x = vf[4 * r + j];
                    hi4[r] = bf_hi(x);
                    lo4[r] = bf_lo(x);
                }
                *(u16x4*)&Bh[n][koff] = hi4;
                *(u16x4*)&Bl[n][koff] = lo4;
            }
        }
        __syncthreads();

        // ---- A fragments: rows rg*32+lr and +16; k runs [8h,8h+8) and [16+8h,+8) ----
        const int ar0 = rg * 32 + lr;
        const int ar1 = ar0 + 16;
        v16bf Ah0 = load_frag(&Ah[ar0][8 * h], &Ah[ar0][16 + 8 * h]);
        v16bf Al0 = load_frag(&Al[ar0][8 * h], &Al[ar0][16 + 8 * h]);
        v16bf Ah1 = load_frag(&Ah[ar1][8 * h], &Ah[ar1][16 + 8 * h]);
        v16bf Al1 = load_frag(&Al[ar1][8 * h], &Al[ar1][16 + 8 * h]);

        #pragma unroll
        for (int nt = 0; nt < 4; ++nt) {
            int n  = cg * 64 + nt * 16 + lr;
            int f  = ((n >> 4) ^ (n >> 6)) & 3;       // uniform per (wave,nt)
            int o0 = (h       ^ f) << 3;              // chunk holding k=8h..8h+7
            int o1 = ((2 + h) ^ f) << 3;              // chunk holding k=16+8h..+7
            v16bf Bhf = load_frag(&Bh[n][o0], &Bh[n][o1]);
            v16bf Blf = load_frag(&Bl[n][o0], &Bl[n][o1]);
            acc[0][nt] = WMMA_BF16(Ah0, Bhf, acc[0][nt]);
            acc[0][nt] = WMMA_BF16(Ah0, Blf, acc[0][nt]);
            acc[0][nt] = WMMA_BF16(Al0, Bhf, acc[0][nt]);
            acc[1][nt] = WMMA_BF16(Ah1, Bhf, acc[1][nt]);
            acc[1][nt] = WMMA_BF16(Ah1, Blf, acc[1][nt]);
            acc[1][nt] = WMMA_BF16(Al1, Bhf, acc[1][nt]);
        }
        __syncthreads();
    }

    // ---- epilogue: C VGPR r <-> row r + 8h, lane lr <-> col ----
    #pragma unroll
    for (int ri = 0; ri < 2; ++ri) {
        #pragma unroll
        for (int nt = 0; nt < 4; ++nt) {
            int col = bCol + cg * 64 + nt * 16 + lr;
            float bv = bias[col];
            #pragma unroll
            for (int r = 0; r < 8; ++r) {
                int row = bRow + rg * 32 + ri * 16 + h * 8 + r;
                float val = acc[ri][nt][r] + bv;
                if (do_relu) val = fmaxf(val, 0.0f);
                C[(size_t)row * N + col] = val;
            }
        }
    }
}

// mixed[j,b,d] = sum_i g[j,b,i] * P[i,b,d]   (g = g_layers[li], shape (M,B,M))
__global__ __launch_bounds__(256)
void mix_kernel(const float* __restrict__ P, const float* __restrict__ g,
                float* __restrict__ mixed)
{
    const int b   = blockIdx.x;
    const int tid = threadIdx.x;
    __shared__ float gs[M_][M_];
    if (tid < M_ * M_) {
        int j = tid / M_, i = tid % M_;
        gs[j][i] = g[((size_t)j * BB + b) * M_ + i];
    }
    __syncthreads();
    for (int d = tid; d < DD; d += 256) {
        float p[M_];
        #pragma unroll
        for (int i = 0; i < M_; ++i) p[i] = P[((size_t)i * BB + b) * DD + d];
        #pragma unroll
        for (int j = 0; j < M_; ++j) {
            float a = 0.f;
            #pragma unroll
            for (int i = 0; i < M_; ++i) a += gs[j][i] * p[i];
            mixed[((size_t)j * BB + b) * DD + d] = a;
        }
    }
}

// feat[b,:] = sum_i g_end[t[b],b,i] * P[i,b,:];  out[b,:] = feat @ W_end[t[b]] + b_end[t[b]]
__global__ __launch_bounds__(256)
void final_kernel(const float* __restrict__ P, const float* __restrict__ g_end,
                  const float* __restrict__ W_end, const float* __restrict__ b_end,
                  const int* __restrict__ t, float* __restrict__ out)
{
    const int b   = blockIdx.x;
    const int tid = threadIdx.x;
    __shared__ float feat[DD];
    __shared__ float gw[M_];
    const int tb = t[b];
    if (tid < M_) gw[tid] = g_end[((size_t)tb * BB + b) * M_ + tid];
    __syncthreads();
    for (int d = tid; d < DD; d += 256) {
        float a = 0.f;
        #pragma unroll
        for (int i = 0; i < M_; ++i) a += gw[i] * P[((size_t)i * BB + b) * DD + d];
        feat[d] = a;
    }
    __syncthreads();
    const float* Wt = W_end + (size_t)tb * DD * DOUT;   // W_end (8 MB) lives in L2
    float acc = b_end[(size_t)tb * DOUT + tid];
    for (int d = 0; d < DD; ++d) acc += feat[d] * Wt[(size_t)d * DOUT + tid];
    out[(size_t)b * DOUT + tid] = acc;
}

extern "C" void kernel_launch(void* const* d_in, const int* in_sizes, int n_in,
                              void* d_out, int out_size, void* d_ws, size_t ws_size,
                              hipStream_t stream) {
    const float* x        = (const float*)d_in[0];   // (B, DIN)
    const float* g_layers = (const float*)d_in[1];   // (L, M, B, M)
    const float* g_end    = (const float*)d_in[2];   // (T, B, M)
    const float* W_start  = (const float*)d_in[3];   // (M, DIN, D)
    const float* b_start  = (const float*)d_in[4];   // (M, D)
    const float* W_mod    = (const float*)d_in[5];   // (L, M, D, D)
    const float* b_mod    = (const float*)d_in[6];   // (L, M, D)
    const float* W_end    = (const float*)d_in[7];   // (T, D, DOUT)
    const float* b_end    = (const float*)d_in[8];   // (T, DOUT)
    const int*   t        = (const int*)d_in[9];     // (B)

    const size_t planeElems = (size_t)M_ * BB * DD;  // 6*1024*1024
    float* P  = (float*)d_ws;                        // (M, B, D)
    float* Mx = P + planeElems;                      // (M, B, D)

    dim3 blk(256);
    dim3 grid(BB / BM, DD / BN, M_);                 // (8, 8, 6)

    // P = x @ W_start[m] + b_start[m]   (x broadcast over m via Astride=0)
    gemm_bf16x3<<<grid, blk, 0, stream>>>(
        x, (size_t)0,
        W_start, (size_t)DIN * DD,
        b_start, (size_t)DD,
        P, (size_t)BB * DD,
        DIN, DD, /*relu=*/0);

    for (int li = 0; li < L_; ++li) {
        mix_kernel<<<dim3(BB), blk, 0, stream>>>(
            P, g_layers + (size_t)li * M_ * BB * M_, Mx);
        // P = relu(mixed @ W_mod[li][j] + b_mod[li][j])
        gemm_bf16x3<<<grid, blk, 0, stream>>>(
            Mx, (size_t)BB * DD,
            W_mod + (size_t)li * M_ * DD * DD, (size_t)DD * DD,
            b_mod + (size_t)li * M_ * DD, (size_t)DD,
            P, (size_t)BB * DD,
            DD, DD, /*relu=*/1);
    }

    final_kernel<<<dim3(BB), blk, 0, stream>>>(
        P, g_end, W_end, b_end, t, (float*)d_out);
}
